// FNO1d_layer_ortho_45973329936886
// MI455X (gfx1250) — matrix operand order repaired and checked by reading, not compiled, not run on hardware
//
#include <hip/hip_runtime.h>

// FNO1d forward for MI455X (gfx1250, wave32, WMMA f32<-f16, TDM staging).
// B=32, S=8192, CIN=2, COUT=1, W=64, M=16, P=24.

#define SLEN 8192
#define BATCH 32
#define WCH 64
#define MMODES 16

typedef __attribute__((ext_vector_type(16))) _Float16 v16h;
typedef __attribute__((ext_vector_type(8)))  float    v8f;
typedef __attribute__((ext_vector_type(4)))  unsigned int v4u;
typedef __attribute__((ext_vector_type(8)))  int v8i;
typedef __attribute__((ext_vector_type(4)))  int v4i;

__device__ __forceinline__ v8f wmma16(v16h a, v16h b, v8f c) {
  // 8 args: (neg_a, A, neg_b, B, c_mod, C, reuse_a, reuse_b)
  return __builtin_amdgcn_wmma_f32_16x16x32_f16(false, a, false, b, (short)0, c,
                                                false, false);
}

// ---- WMMA fragment loaders (contiguous k order k = 16*(lane/16) + j, same in A and B) ----
// A (16x32 f16, MxK): lane L (L<16) row M=L holds K=0..15,
// lane L+16 same row holds K=16..31.
__device__ __forceinline__ v16h load_a_row(const _Float16* A, int lda, int lane) {
  v16h f;
  const _Float16* row = A + (lane & 15) * lda;
  int hi8 = (lane & 16) >> 1;  // 0 | 8
#pragma unroll
  for (int j = 0; j < 16; ++j) {
    int k = (hi8 << 1) + j;
    f[j] = row[k];
  }
  return f;
}
// A from column-major storage: element (r,k) at A[r + k*ldk]
__device__ __forceinline__ v16h load_a_col(const _Float16* A, long ldk, int lane) {
  v16h f;
  int r = lane & 15;
  int hi8 = (lane & 16) >> 1;
#pragma unroll
  for (int j = 0; j < 16; ++j) {
    int k = (hi8 << 1) + j;
    f[j] = A[r + (long)k * ldk];
  }
  return f;
}
// B (32x16 f16, KxN): lane L col N=L&15; lanes 0-15 K=0..15, lanes 16-31 K=16..31.
// "n-major" storage: element (k,n) at B[n*ldn + k]  (contiguous in k per lane)
__device__ __forceinline__ v16h load_b_nmaj(const _Float16* B, int ldn, int lane) {
  v16h f;
  const _Float16* p = B + (lane & 15) * ldn + (lane & 16);
#pragma unroll
  for (int j = 0; j < 16; ++j) f[j] = p[j];
  return f;
}
// row-major KxN storage: element (k,n) at B[k*ldb + n]
__device__ __forceinline__ v16h load_b_rmaj(const _Float16* B, long ldb, int lane) {
  v16h f;
  const _Float16* p = B + (long)(lane & 16) * ldb + (lane & 15);
#pragma unroll
  for (int j = 0; j < 16; ++j) f[j] = p[(long)j * ldb];
  return f;
}

// ---- Tensor Data Mover: DMA a dense 64-row x 128-half tile (16 KB) of the
// activation tensor H (row stride SLEN halves) into LDS.  D# packing per
// CDNA5 ISA.  This toolchain uses the 6-arg
// builtin form: (v4u g0, v8i g1, v4i, v4i, v8i, i32 cpol); groups 2/3 and the
// trailing group are unused for a 2D tensor -> zero-filled.
__device__ __forceinline__ void tdm_load_64x128(const _Float16* gsrc,
                                                _Float16* lds_dst) {
  unsigned long long ga = (unsigned long long)(size_t)gsrc;
  unsigned lo = (unsigned)(size_t)lds_dst;  // shared aperture low 32b == LDS byte offset
  v4u g0;
  g0[0] = 1u;                                               // count=1 (user D#)
  g0[1] = lo;                                               // lds_addr (bytes)
  g0[2] = (unsigned)(ga & 0xFFFFFFFFu);                     // global_addr[31:0]
  g0[3] = (unsigned)((ga >> 32) & 0x01FFFFFFu) | (2u << 30); // addr[56:32] | type=2
  v8i g1;
  g1[0] = (int)(1u << 16);      // workgroup_mask=0 | data_size=1 (2 bytes)
  g1[1] = (int)(8192u << 16);   // tensor_dim0 = 8192 (bits 79:48, low half)
  g1[2] = (int)(64u << 16);     // tensor_dim0 hi=0 | tensor_dim1 = 64
  g1[3] = (int)(128u << 16);    // tensor_dim1 hi=0 | tile_dim0 = 128
  g1[4] = 64;                   // tile_dim1 = 64, tile_dim2 = 0 (2D tile)
  g1[5] = 8192;                 // tensor_dim0_stride = 8192 elements
  g1[6] = 0;                    // stride hi | tensor_dim1_stride lo
  g1[7] = 0;
  v4i z4 = {0, 0, 0, 0};        // groups 2/3 unused (<=2D tensor)
  v8i z8 = {0, 0, 0, 0, 0, 0, 0, 0};
  __builtin_amdgcn_tensor_load_to_lds(g0, g1, z4, z4, z8, 0);
}

// ---- Kernels ----

// Fourier basis tables (f16): Tn[n][s] (n-major, analysis B) and Ts[s][c]
// (s-major, synthesis B). Rows/cols 0..15 = cos(2*pi*k*s/S), 16..31 = sin.
__global__ void k_basis(_Float16* Tn, _Float16* Ts) {
  int s = blockIdx.x * blockDim.x + threadIdx.x;
  if (s >= SLEN) return;
#pragma unroll 4
  for (int k = 0; k < MMODES; ++k) {
    int ph = (k * s) & (SLEN - 1);  // reduce phase for f32 accuracy
    float ang = 6.2831853071795864769f * (float)ph * (1.0f / (float)SLEN);
    float si, co;
    __sincosf(ang, &si, &co);
    Tn[k * SLEN + s]        = (_Float16)co;
    Tn[(16 + k) * SLEN + s] = (_Float16)si;
    Ts[s * 32 + k]          = (_Float16)co;
    Ts[s * 32 + 16 + k]     = (_Float16)si;
  }
}

// f16 copies of small weights + fused tc matrix atc[p][c]=ica_w[p,c]*fc2_w[c]
// (padded to 32 rows with zeros so tc GEMM runs as two 16-col WMMA tiles).
__global__ void k_convert(const float* pw, const float* fc1, const float* fc2,
                          const float* ica, _Float16* pwf, _Float16* fc1f,
                          _Float16* atc) {
  int t = blockIdx.x * blockDim.x + threadIdx.x;
  if (t < 4 * WCH * WCH) pwf[t] = (_Float16)pw[t];
  if (t < 128 * WCH)     fc1f[t] = (_Float16)fc1[t];
  if (t < 32 * 128) {
    int p = t >> 7, c = t & 127;
    atc[t] = (_Float16)((p < 24) ? ica[p * 128 + c] * fc2[c] : 0.0f);
  }
}

// fc0 lift: h[b,i,s] = x[b,s,:2] . fc0_w[i,:2] + fc0_b[i]  (store f16, (B*W, S))
__global__ void k_lift(const float* __restrict__ x, const float* __restrict__ w,
                       const float* __restrict__ bb, _Float16* __restrict__ h) {
  long idx = (long)blockIdx.x * blockDim.x + threadIdx.x;  // over B*W*S
  int s = (int)(idx & (SLEN - 1));
  long t = idx >> 13;
  int i = (int)(t & (WCH - 1));
  int b = (int)(t >> 6);
  long xo = ((long)b * SLEN + s) * 2;
  float v = x[xo] * w[i * 2] + x[xo + 1] * w[i * 2 + 1] + bb[i];
  h[idx] = (_Float16)v;
}

// Mode analysis GEMM: CS(2048 x 32) = H(2048 x 8192) x basis(8192 x 32).
// cols 0..15 = sum h*cos (Re xf), cols 16..31 = sum h*sin (= -Im xf).
__global__ void k_analysis(const _Float16* __restrict__ h,
                           const _Float16* __restrict__ Tn,
                           float* __restrict__ CS) {
  int wid = (blockIdx.x * blockDim.x + threadIdx.x) >> 5;  // 256 waves
  int lane = threadIdx.x & 31;
  int mt = wid >> 1;       // 128 row tiles of (b,i)
  int nt = wid & 1;        // 2 col tiles
  const _Float16* Arow = h + (long)mt * 16 * SLEN;
  const _Float16* Bn = Tn + nt * 16 * SLEN;
  v8f acc = {};
  for (int ks = 0; ks < SLEN; ks += 32) {
    __builtin_prefetch(Arow + ks + 512, 0, 1);
    v16h a = load_a_row(Arow + ks, SLEN, lane);
    v16h b = load_b_nmaj(Bn + ks, SLEN, lane);
    acc = wmma16(a, b, acc);
  }
  int n = nt * 16 + (lane & 15);
  int mb = (lane & 16) >> 1;
#pragma unroll
  for (int r = 0; r < 8; ++r)
    CS[(long)(mt * 16 + mb + r) * 32 + n] = acc[r];
}

// Complex mode mixing (tiny, f32 VALU) + fold irfft scale factors:
// of[b,o,k] = sum_i xf[b,i,k]*w[i,o,k];  xf = C - i*S
// AB[.,k]    = (k==0?1:2)/S * Re(of)   (cos coefficient)
// AB[.,16+k] = (k==0?0:-2/S) * Im(of)  (sin coefficient)
__global__ void k_mix(const float* __restrict__ CS, const float* __restrict__ wr,
                      const float* __restrict__ wi, _Float16* __restrict__ AB) {
  int idx = blockIdx.x * blockDim.x + threadIdx.x;  // B*W*M = 32768
  int k = idx & 15;
  int o = (idx >> 4) & 63;
  int b = idx >> 10;
  float re = 0.f, im = 0.f;
#pragma unroll 4
  for (int i = 0; i < WCH; ++i) {
    float C = CS[(b * 64 + i) * 32 + k];
    float Sv = CS[(b * 64 + i) * 32 + 16 + k];
    float r = wr[(i * 64 + o) * 16 + k];
    float m = wi[(i * 64 + o) * 16 + k];
    re += C * r + Sv * m;   // Re((C - iS)(r + im))
    im += C * m - Sv * r;   // Im
  }
  const float invS = 1.0f / (float)SLEN;
  float a = (k == 0 ? invS : 2.0f * invS) * re;
  float bq = (k == 0 ? 0.0f : -2.0f * invS) * im;
  AB[(b * 64 + o) * 32 + k] = (_Float16)a;
  AB[(b * 64 + o) * 32 + 16 + k] = (_Float16)bq;
}

// Fused irfft-synthesis + pointwise conv + bias + relu, TDM-staged.
// Workgroup = (batch b, 128-wide s block): wave 0 DMAs H[b, 0:64, s0:s0+128]
// (16 KB) into LDS via tensor_load_to_lds; each of the 8 waves then computes
// a 16-wide s subtile for all 4 o-tiles:
//   out(16 (b,o) x 16 s) = AB(16x32) x Ts(32x16) + pw(16x64) x H_lds(64x16)
__global__ void k_synth(const _Float16* __restrict__ AB,
                        const _Float16* __restrict__ Ts,
                        const _Float16* __restrict__ hcur,
                        const _Float16* __restrict__ pwf,
                        const float* __restrict__ pwb,
                        _Float16* __restrict__ hnext, int dorelu) {
  __shared__ _Float16 Hs[WCH * 128];  // 16 KB activation tile
  int lane = threadIdx.x & 31;
  int wv = threadIdx.x >> 5;
  int b = blockIdx.x >> 6;            // 32 batches
  int s0 = (blockIdx.x & 63) << 7;    // 64 s-blocks of 128

  if (threadIdx.x < 32) {             // one TDM descriptor per workgroup
    tdm_load_64x128(hcur + (long)b * WCH * SLEN + s0, Hs);
    __builtin_amdgcn_s_wait_tensorcnt(0);
  }
  __syncthreads();

  int sl0 = wv << 4;                  // this wave's s subtile within the block
  int n = lane & 15;
  int mb = (lane & 16) >> 1;
  v16h bts = load_b_nmaj(Ts + (s0 + sl0) * 32, 32, lane);
#pragma unroll
  for (int ot = 0; ot < 4; ++ot) {
    v8f acc = {};
    acc = wmma16(load_a_row(AB + ((b << 2) + ot) * 16 * 32, 32, lane), bts, acc);
#pragma unroll
    for (int kk = 0; kk < WCH; kk += 32) {
      acc = wmma16(load_a_row(pwf + ot * 16 * 64 + kk, 64, lane),
                   load_b_rmaj(Hs + kk * 128 + sl0, 128, lane), acc);
    }
#pragma unroll
    for (int r = 0; r < 8; ++r) {
      int row = b * 64 + ot * 16 + mb + r;  // (b,o) flat
      float v = acc[r] + pwb[row & 63];
      if (dorelu) v = v > 0.0f ? v : 0.0f;
      hnext[(long)row * SLEN + s0 + sl0 + n] = (_Float16)v;
    }
  }
}

// Head: TDM-stage H[b,0:64,sblock], fc1 + relu (WMMA, D-layout -> LDS ->
// A-layout), tc GEMM vs atc (128x32 padded) + ica_b, store tc and out=sum_p tc.
__global__ void k_head(const _Float16* __restrict__ h,
                       const _Float16* __restrict__ fc1f,
                       const float* __restrict__ fc1b,
                       const _Float16* __restrict__ atc,
                       const float* __restrict__ icab,
                       float* __restrict__ outp, float* __restrict__ tcp) {
  __shared__ _Float16 Hs[WCH * 128];          // 16 KB activation tile
  __shared__ _Float16 ldsbuf[8 * 16 * 128];   // 32 KB fc1 tiles (1 per wave)
  int wv = threadIdx.x >> 5;
  int lane = threadIdx.x & 31;
  int b = blockIdx.x >> 6;
  int s0 = (blockIdx.x & 63) << 7;
  int sl0 = wv << 4;
  long R0 = (long)b * SLEN + s0 + sl0;        // 16 consecutive (b,s) rows
  _Float16* lw = ldsbuf + wv * (16 * 128);

  if (threadIdx.x < 32) {
    tdm_load_64x128(h + (long)b * WCH * SLEN + s0, Hs);
    __builtin_amdgcn_s_wait_tensorcnt(0);
  }
  __syncthreads();

  int n = lane & 15;
  int mb = (lane & 16) >> 1;

  // fc1: A = H^T tile (16 rows (b,s) x 64 i), column-major in LDS (ldk=128).
  v16h a0 = load_a_col(Hs + sl0, 128, lane);
  v16h a1 = load_a_col(Hs + sl0 + 32 * 128, 128, lane);
#pragma unroll
  for (int nt = 0; nt < 8; ++nt) {
    v8f acc = {};
    acc = wmma16(a0, load_b_nmaj(fc1f + nt * 16 * 64, 64, lane), acc);
    acc = wmma16(a1, load_b_nmaj(fc1f + nt * 16 * 64 + 32, 64, lane), acc);
    float bias = fc1b[nt * 16 + n];
#pragma unroll
    for (int r = 0; r < 8; ++r) {
      float v = acc[r] + bias;
      v = v > 0.0f ? v : 0.0f;                       // relu
      lw[(mb + r) * 128 + nt * 16 + n] = (_Float16)v;
    }
  }
  __syncthreads();

  // tc: (16 x 128) x (128 x 32)
  v8f t0 = {}, t1 = {};
#pragma unroll
  for (int kk = 0; kk < 4; ++kk) {
    v16h a = load_a_row(lw + kk * 32, 128, lane);
    t0 = wmma16(a, load_b_nmaj(atc + kk * 32, 128, lane), t0);
    t1 = wmma16(a, load_b_nmaj(atc + 16 * 128 + kk * 32, 128, lane), t1);
  }
  float bias0 = icab[n];
  float bias1 = (n < 8) ? icab[16 + n] : 0.0f;
#pragma unroll
  for (int r = 0; r < 8; ++r) {
    long R = R0 + mb + r;
    float v0 = t0[r] + bias0;
    tcp[R * 24 + n] = v0;
    float rowp = v0;
    if (n < 8) {
      float v1 = t1[r] + bias1;
      tcp[R * 24 + 16 + n] = v1;
      rowp += v1;
    }
    rowp += __shfl_xor(rowp, 1);   // reduce over the 16 lanes holding this row
    rowp += __shfl_xor(rowp, 2);
    rowp += __shfl_xor(rowp, 4);
    rowp += __shfl_xor(rowp, 8);
    if (n == 0) outp[R] = rowp;
  }
}

extern "C" void kernel_launch(void* const* d_in, const int* in_sizes, int n_in,
                              void* d_out, int out_size, void* d_ws, size_t ws_size,
                              hipStream_t stream) {
  const float* x       = (const float*)d_in[0];
  const float* fc0_w   = (const float*)d_in[1];
  const float* fc0_b   = (const float*)d_in[2];
  const float* conv_wr = (const float*)d_in[3];
  const float* conv_wi = (const float*)d_in[4];
  const float* pw_w    = (const float*)d_in[5];
  const float* pw_b    = (const float*)d_in[6];
  const float* fc1_w   = (const float*)d_in[7];
  const float* fc1_b   = (const float*)d_in[8];
  const float* fc2_w   = (const float*)d_in[9];
  const float* ica_w   = (const float*)d_in[10];
  const float* ica_b   = (const float*)d_in[11];

  float* outp = (float*)d_out;                 // (B,S,1) = 262144
  float* tcp  = outp + (long)BATCH * SLEN;     // (B,S,1,24)

  // workspace carve (all chunks power-of-two sized => aligned)
  char* ws = (char*)d_ws;
  const size_t HS = (size_t)BATCH * WCH * SLEN;          // 16.78M elems
  _Float16* h0  = (_Float16*)ws; ws += HS * 2;
  _Float16* h1  = (_Float16*)ws; ws += HS * 2;
  _Float16* Tn  = (_Float16*)ws; ws += (size_t)32 * SLEN * 2;
  _Float16* Ts  = (_Float16*)ws; ws += (size_t)SLEN * 32 * 2;
  float*    CS  = (float*)ws;    ws += (size_t)2048 * 32 * 4;
  _Float16* AB  = (_Float16*)ws; ws += (size_t)2048 * 32 * 2;
  _Float16* pwf = (_Float16*)ws; ws += (size_t)4 * 64 * 64 * 2;
  _Float16* f1f = (_Float16*)ws; ws += (size_t)128 * 64 * 2;
  _Float16* atc = (_Float16*)ws; ws += (size_t)32 * 128 * 2;
  (void)ws_size; (void)in_sizes; (void)n_in; (void)out_size;

  k_basis<<<32, 256, 0, stream>>>(Tn, Ts);
  k_convert<<<64, 256, 0, stream>>>(pw_w, fc1_w, fc2_w, ica_w, pwf, f1f, atc);
  k_lift<<<65536, 256, 0, stream>>>(x, fc0_w, fc0_b, h0);

  _Float16* hc = h0;
  _Float16* hn = h1;
  for (int l = 0; l < 4; ++l) {
    k_analysis<<<32, 256, 0, stream>>>(hc, Tn, CS);
    k_mix<<<128, 256, 0, stream>>>(CS, conv_wr + (long)l * 65536,
                                   conv_wi + (long)l * 65536, AB);
    k_synth<<<2048, 256, 0, stream>>>(AB, Ts, hc, pwf + l * 4096, pw_b + l * 64,
                                      hn, l < 3 ? 1 : 0);
    _Float16* t = hc; hc = hn; hn = t;
  }
  k_head<<<2048, 256, 0, stream>>>(hc, f1f, fc1_b, atc, ica_b, outp, tcp);
}
